// SimpleRNNClassifier_37709812859259
// MI455X (gfx1250) — matrix operand order repaired and checked by reading, not compiled, not run on hardware
//
#include <hip/hip_runtime.h>
#include <hip/hip_bf16.h>
#include <stdint.h>

// ---------------------------------------------------------------------------
// GRU classifier: T=16384 steps, I=H=2048, O=1000.
// Phase 1: igates = x @ W_ih^T + bias  -> bf16 WMMA GEMM (fp32 accum)
// Phase 2: sequential GRU scan          -> persistent kernel, grid barrier/step
//          W_hh kept in bf16: halves the dominant per-step L2 traffic
//          (16384 x 25 MB instead of 16384 x 50 MB)
// Phase 3: out = W_out @ h_T + b + bx   -> wave-per-row matvec
// ---------------------------------------------------------------------------

#define T_STEPS 16384
#define I_DIM   2048
#define H_DIM   2048
#define G3      (3 * H_DIM)   // 6144
#define O_DIM   1000

typedef __bf16 v16bf __attribute__((ext_vector_type(16)));
typedef __bf16 v8bf  __attribute__((ext_vector_type(8)));
typedef float  v8f   __attribute__((ext_vector_type(8)));

// ---------------------------------------------------------------------------
// fp32 -> bf16 conversion (RNE via clang cast)
// ---------------------------------------------------------------------------
__global__ void cvt_f32_bf16(const float* __restrict__ s, __bf16* __restrict__ d,
                             size_t n) {
    size_t i = (size_t)blockIdx.x * blockDim.x + threadIdx.x;
    size_t stride = (size_t)gridDim.x * blockDim.x;
    for (; i < n; i += stride) d[i] = (__bf16)s[i];
}

// ---------------------------------------------------------------------------
// WMMA fragment loaders (wave32; contiguous k = 16*(lane/16) + s, same in A and B)
// ---------------------------------------------------------------------------
// A 16x32 bf16: lane<16 -> row=lane,   K=k0..k0+15 contiguous
//               lane>=16 -> row=lane-16, K=k0+16..k0+31 contiguous
__device__ __forceinline__ v16bf load_frag_a(const __bf16* __restrict__ base,
                                             int ld, int k0) {
    int lane = threadIdx.x & 31;
    const __bf16* p = base + (size_t)(lane & 15) * ld + k0 + ((lane >> 4) << 4);
    v8bf c0 = *(const v8bf*)(p);
    v8bf c1 = *(const v8bf*)(p + 8);
    v16bf r;
#pragma unroll
    for (int i = 0; i < 8; ++i) { r[i] = c0[i]; r[i + 8] = c1[i]; }
    return r;
}

// B 32x16 bf16 (sourced from row-major W[col][k]):
//   lane<16 -> col=lane, K=k0..k0+15 contiguous; lane>=16 -> K=k0+16..k0+31
__device__ __forceinline__ v16bf load_frag_b(const __bf16* __restrict__ base,
                                             int ld, int k0) {
    int lane = threadIdx.x & 31;
    const __bf16* p = base + (size_t)(lane & 15) * ld + k0 + ((lane >> 4) << 4);
    return *(const v16bf*)p;   // 32B contiguous, 32B aligned
}

__device__ __forceinline__ v8f wmma_bf16(v16bf a, v16bf b, v8f c) {
    return __builtin_amdgcn_wmma_f32_16x16x32_bf16(
        /*neg_a=*/false, a, /*neg_b=*/false, b,
        /*c_mod=*/(short)0, c, /*reuse_a=*/false, /*reuse_b=*/false);
}

// ---------------------------------------------------------------------------
// GEMM: igates[T x 6144] = Xb[T x 2048] * Wb^T (Wb is [6144 x 2048]) + bias
// Block tile 128x64 (8 waves, each wave a 32x32 tile = 2x2 WMMA tiles)
// grid = (6144/64, 16384/128) = (96, 128)
// ---------------------------------------------------------------------------
__global__ __launch_bounds__(256)
void igates_gemm(const __bf16* __restrict__ X, const __bf16* __restrict__ W,
                 const float* __restrict__ bias, float* __restrict__ ig) {
    const int wave = threadIdx.x >> 5;
    const int lane = threadIdx.x & 31;
    const int wrow = wave >> 1;       // 0..3
    const int wcol = wave & 1;        // 0..1
    const int row0 = blockIdx.y * 128 + wrow * 32;
    const int col0 = blockIdx.x * 64 + wcol * 32;

    const __bf16* Xa0 = X + (size_t)row0 * I_DIM;
    const __bf16* Xa1 = X + (size_t)(row0 + 16) * I_DIM;
    const __bf16* Wb0 = W + (size_t)col0 * I_DIM;
    const __bf16* Wb1 = W + (size_t)(col0 + 16) * I_DIM;

    v8f acc00 = {}, acc01 = {}, acc10 = {}, acc11 = {};

    for (int k = 0; k < I_DIM; k += 32) {
        v16bf a0 = load_frag_a(Xa0, I_DIM, k);
        v16bf a1 = load_frag_a(Xa1, I_DIM, k);
        v16bf b0 = load_frag_b(Wb0, I_DIM, k);
        v16bf b1 = load_frag_b(Wb1, I_DIM, k);
        if (k + 32 < I_DIM) {  // stream next K-slab into cache
            __builtin_prefetch(Xa0 + (size_t)(lane & 15) * I_DIM + k + 32, 0, 1);
            __builtin_prefetch(Wb0 + (size_t)(lane & 15) * I_DIM + k + 32, 0, 1);
        }
        acc00 = wmma_bf16(a0, b0, acc00);
        acc01 = wmma_bf16(a0, b1, acc01);
        acc10 = wmma_bf16(a1, b0, acc10);
        acc11 = wmma_bf16(a1, b1, acc11);
    }

    // C/D layout: VGPR r -> row r (lanes 0-15) / row r+8 (lanes 16-31)
    const int n  = lane & 15;
    const int m0 = (lane >> 4) * 8;
    const float bc0 = bias[col0 + n];
    const float bc1 = bias[col0 + 16 + n];
#pragma unroll
    for (int r = 0; r < 8; ++r) {
        ig[(size_t)(row0 + m0 + r) * G3 + col0 + n]           = acc00[r] + bc0;
        ig[(size_t)(row0 + m0 + r) * G3 + col0 + 16 + n]      = acc01[r] + bc1;
        ig[(size_t)(row0 + 16 + m0 + r) * G3 + col0 + n]      = acc10[r] + bc0;
        ig[(size_t)(row0 + 16 + m0 + r) * G3 + col0 + 16 + n] = acc11[r] + bc1;
    }
}

// ---------------------------------------------------------------------------
// Persistent GRU scan. grid = 128 blocks x 256 threads; block owns 16 hidden
// units; wave owns 2; per unit: 3 dot products of length 2048 against h_old
// (staged in LDS, read as float2 / ds_load_b64), weights read as packed bf16
// pairs (one u32 load -> two FMAs, bf16->f32 is a shift). Gate update is
// block-local. Grid-wide barrier per timestep via monotonic atomic counter
// + s_sleep spin.
// ---------------------------------------------------------------------------
__device__ __forceinline__ float wave_reduce(float v) {
#pragma unroll
    for (int off = 16; off > 0; off >>= 1) v += __shfl_down(v, off, 32);
    return v;
}
__device__ __forceinline__ float sigmoidf(float x) {
    return 1.0f / (1.0f + __expf(-x));
}

__global__ __launch_bounds__(256)
void gru_scan(const uint32_t* __restrict__ whb,   // W_hh as packed bf16 pairs
              const float* __restrict__ igates,
              const float* __restrict__ bias_n, float* __restrict__ hbuf,
              unsigned int* __restrict__ counter) {
    __shared__ float hs[H_DIM];
    const int lane = threadIdx.x & 31;
    const int wave = threadIdx.x >> 5;
    const int jb   = blockIdx.x * 16;
    const unsigned int nb = gridDim.x;
    const int HP = H_DIM / 2;   // 1024 bf16 pairs per row

    for (int t = 0; t < T_STEPS; ++t) {
        const float* hold = hbuf + (t & 1) * H_DIM;
        float*       hnew = hbuf + ((t + 1) & 1) * H_DIM;
        const float* ig   = igates + (size_t)t * G3;

        // stage h_old into LDS
        for (int k = threadIdx.x; k < H_DIM; k += 256) hs[k] = hold[k];
        __syncthreads();

#pragma unroll
        for (int jj = 0; jj < 2; ++jj) {
            const int j = jb + wave * 2 + jj;
            float d[3];
#pragma unroll
            for (int g = 0; g < 3; ++g) {
                const uint32_t* row = whb + (size_t)(g * H_DIM + j) * HP;
                float s = 0.0f;
#pragma unroll 4
                for (int k2 = lane; k2 < HP; k2 += 32) {
                    uint32_t w2 = row[k2];                       // 2 x bf16
                    float2 h2 = *(const float2*)&hs[2 * k2];     // ds_load_b64
                    float wlo = __uint_as_float(w2 << 16);
                    float whi = __uint_as_float(w2 & 0xffff0000u);
                    s = fmaf(wlo, h2.x, s);
                    s = fmaf(whi, h2.y, s);
                }
                d[g] = wave_reduce(s);
            }
            if (lane == 0) {
                float r = sigmoidf(ig[j] + d[0]);
                float z = sigmoidf(ig[H_DIM + j] + d[1]);
                float nn = tanhf(ig[2 * H_DIM + j] + r * (d[2] + bias_n[j]));
                hnew[j] = nn + z * (hs[j] - nn);
            }
        }

        // grid barrier (monotonic counter; no reset needed)
        __syncthreads();
        if (threadIdx.x == 0) {
            __threadfence();
            atomicAdd(counter, 1u);
            const unsigned int target = nb * (unsigned int)(t + 1);
            while (__hip_atomic_load(counter, __ATOMIC_ACQUIRE,
                                     __HIP_MEMORY_SCOPE_AGENT) < target) {
                __builtin_amdgcn_s_sleep(2);
            }
            __threadfence();
        }
        __syncthreads();
    }
}

// ---------------------------------------------------------------------------
// out = W_out @ h + b_out + bias_extra ; one wave per output row
// ---------------------------------------------------------------------------
__global__ __launch_bounds__(256)
void out_proj(const float* __restrict__ w, const float* __restrict__ h,
              const float* __restrict__ b, const float* __restrict__ bx,
              float* __restrict__ out) {
    const int o = blockIdx.x * 8 + (threadIdx.x >> 5);
    const int lane = threadIdx.x & 31;
    if (o >= O_DIM) return;
    const float* row = w + (size_t)o * H_DIM;
    float s = 0.0f;
#pragma unroll 4
    for (int k = lane; k < H_DIM; k += 32) s += row[k] * h[k];
    s = wave_reduce(s);
    if (lane == 0) out[o] = s + b[o] + bx[o];
}

// ---------------------------------------------------------------------------
// launch
// ---------------------------------------------------------------------------
extern "C" void kernel_launch(void* const* d_in, const int* in_sizes, int n_in,
                              void* d_out, int out_size, void* d_ws, size_t ws_size,
                              hipStream_t stream) {
    const float* x      = (const float*)d_in[0];
    const float* wih    = (const float*)d_in[1];
    const float* whh    = (const float*)d_in[2];
    const float* bias   = (const float*)d_in[3];
    const float* bias_n = (const float*)d_in[4];
    const float* wout   = (const float*)d_in[5];
    const float* bout   = (const float*)d_in[6];
    const float* bx     = (const float*)d_in[7];
    float* out = (float*)d_out;

    char* ws = (char*)d_ws;
    // workspace layout (all 256B aligned)
    const size_t oXB  = 0;                                   // T*I bf16   = 64 MB
    const size_t oWB  = oXB + (size_t)T_STEPS * I_DIM * 2;   // 3H*I bf16  = 24 MB
    const size_t oIG  = oWB + (size_t)G3 * I_DIM * 2;        // T*3H fp32  = 384 MB
    const size_t oH   = oIG + (size_t)T_STEPS * G3 * 4;      // 2*H fp32   = 16 KB
    const size_t oCNT = oH + 2 * (size_t)H_DIM * 4;          // counter (256 B)
    const size_t oWHB = oCNT + 256;                          // 3H*H bf16  = 24 MB

    __bf16* xb  = (__bf16*)(ws + oXB);
    __bf16* wb  = (__bf16*)(ws + oWB);
    float*  ig  = (float*)(ws + oIG);
    float*  hb  = (float*)(ws + oH);
    unsigned int* cnt = (unsigned int*)(ws + oCNT);
    __bf16* whb = (__bf16*)(ws + oWHB);

    // init: h0 = 0 (both ping-pong halves), barrier counter = 0
    hipMemsetAsync(hb, 0, 2 * (size_t)H_DIM * sizeof(float), stream);
    hipMemsetAsync(cnt, 0, 256, stream);

    // Phase 0: fp32 -> bf16 for GEMM + recurrent weights
    cvt_f32_bf16<<<4096, 256, 0, stream>>>(x, xb, (size_t)T_STEPS * I_DIM);
    cvt_f32_bf16<<<4096, 256, 0, stream>>>(wih, wb, (size_t)G3 * I_DIM);
    cvt_f32_bf16<<<4096, 256, 0, stream>>>(whh, whb, (size_t)G3 * H_DIM);

    // Phase 1: igates GEMM (WMMA bf16 -> fp32)
    dim3 ggrid(G3 / 64, T_STEPS / 128);
    igates_gemm<<<ggrid, 256, 0, stream>>>(xb, wb, bias, ig);

    // Phase 2: persistent sequential scan (128 co-resident blocks)
    gru_scan<<<128, 256, 0, stream>>>((const uint32_t*)whb, ig, bias_n, hb, cnt);

    // Phase 3: output projection; final h lands in ping-pong half 0 (T even)
    out_proj<<<(O_DIM + 7) / 8, 256, 0, stream>>>(wout, hb, bout, bx, out);
}
